// QuantumEntanglementNetwork_89154931130741
// MI455X (gfx1250) — matrix operand order repaired and checked by reading, not compiled, not run on hardware
//
#include <hip/hip_runtime.h>
#include <cstdint>
#include <cstddef>

// ---------------------------------------------------------------------------
// QuantumEntanglementNetwork for MI455X (gfx1250, wave32, WMMA)
//
// Heavy math: per pair p: H = tanh(X@W1+b1); Bell = H@W2+b2, X = [8192 x 256]
// Lowered to f16 WMMA (v_wmma_f32_16x16x32_f16) with f32 accumulation.
// Sequential layer update unrolled to closed-form linear coefficients.
// ---------------------------------------------------------------------------

typedef __attribute__((ext_vector_type(16))) _Float16 v16h;
typedef __attribute__((ext_vector_type(8)))  _Float16 v8h;
typedef __attribute__((ext_vector_type(4)))  _Float16 v4h;
typedef __attribute__((ext_vector_type(8)))  float    v8f;

#define NL 8
#define ND 128
#define NB 8192
#define NP 28
#define TWO_D 256
#define UPD_ELEMS (NL*NB*ND)   // 8,388,608 floats of "updated" in d_out
#define XS 264                 // LDS row stride in halves (528B: 4-bank rotate)

__constant__ unsigned char cII[NP] = {0,0,0,0,0,0,0,1,1,1,1,1,1,2,2,2,2,2,3,3,3,3,4,4,4,5,5,6};
__constant__ unsigned char cJJ[NP] = {1,2,3,4,5,6,7,2,3,4,5,6,7,3,4,5,6,7,4,5,6,7,5,6,7,6,7,7};
// 0.9^e
__constant__ float cPOW9[8] = {1.0f,0.9f,0.81f,0.729f,0.6561f,0.59049f,0.531441f,0.4782969f};

union V16u { v16h v; v8h h[2]; };

__device__ __forceinline__ uint32_t hash_u32(uint32_t x) {
  x ^= x >> 16; x *= 0x7feb352dU;
  x ^= x >> 15; x *= 0x846ca68bU;
  x ^= x >> 16;
  return x;
}

// deterministic pseudo-normal (sum of 4 uniforms, unit variance)
__device__ __forceinline__ float gaussn(uint32_t s) {
  uint32_t a = hash_u32(s * 0x9E3779B9u + 0x85EBCA6Bu);
  uint32_t b = hash_u32(a ^ 0xC2B2AE35u);
  float u0 = (float)(a & 0xFFFFu) * (1.f/65536.f) - 0.5f;
  float u1 = (float)(a >> 16)     * (1.f/65536.f) - 0.5f;
  float u2 = (float)(b & 0xFFFFu) * (1.f/65536.f) - 0.5f;
  float u3 = (float)(b >> 16)     * (1.f/65536.f) - 0.5f;
  return (u0+u1+u2+u3) * 1.7320508f;
}

// ---- kernel 1: updated-base = 0.9^7 * s ; Xh = (f16)s --------------------
__global__ __launch_bounds__(256) void prep_states(
    const float* __restrict__ s, float* __restrict__ out, _Float16* __restrict__ Xh) {
  int idx = blockIdx.x * 256 + threadIdx.x;              // exactly 2,097,152
  float4 v = ((const float4*)s)[idx];
  const float c = 0.4782969f;                            // 0.9^7
  float4 o; o.x = v.x*c; o.y = v.y*c; o.z = v.z*c; o.w = v.w*c;
  ((float4*)out)[idx] = o;
  v4h hv = {(_Float16)v.x, (_Float16)v.y, (_Float16)v.z, (_Float16)v.w};
  ((v4h*)Xh)[idx] = hv;
}

// ---- kernel 2: per-row rank-1 entropy table SL[L*B] ----------------------
__global__ __launch_bounds__(256) void prep_entropy(
    const float* __restrict__ s, float* __restrict__ SL) {
  int idx = blockIdx.x * 256 + threadIdx.x;              // exactly 65,536 rows
  const float4* r = (const float4*)(s + (size_t)idx * ND);
  float sq = 0.f;
  #pragma unroll 8
  for (int t = 0; t < ND/4; ++t) {
    float4 v = r[t];
    sq += v.x*v.x + v.y*v.y + v.z*v.z + v.w*v.w;
  }
  float dn  = fmaxf(sqrtf(sq), 1e-12f);
  float lam = fmaxf(sq / (dn*dn), 1e-12f);
  // zero_term = -(D-1)*eps*log(eps), log(1e-12) = -27.631021...
  SL[idx] = -lam * __logf(lam) + 127.f * 1e-12f * 27.631021115928547f;
}

// ---- kernel 3: pack W1/W2 into f16 WMMA B-fragment layout ----------------
// Fragment (kb,nb) = 512 halves laid out as [chunk(2)][lane(32)][half(8)]:
//   lane<16 -> base = kb*32, lane>=16 -> base = kb*32+8
//   within a lane-half: half h of chunk c is K = base + c*16 + h; N = nb*16 + lane%16
__global__ __launch_bounds__(256) void pack_weights(
    const float* __restrict__ W1, const float* __restrict__ W2,
    _Float16* __restrict__ WPk) {
  uint32_t idx = blockIdx.x * 256u + threadIdx.x;        // exactly 3,670,016
  uint32_t h  =  idx        & 7u;
  uint32_t l  = (idx >> 3)  & 31u;
  uint32_t c  = (idx >> 8)  & 1u;
  uint32_t nb = (idx >> 9)  & 15u;
  uint32_t kb = (idx >> 13) & 7u;
  uint32_t st = (idx >> 16) & 1u;
  uint32_t p  =  idx >> 17;
  uint32_t k = (kb << 5) + ((l < 16u) ? 0u : 8u) + (c << 4) + h;
  uint32_t n = (nb << 4) + (l & 15u);
  const float* src = (st == 0u) ? W1 : W2;
  WPk[idx] = (_Float16)src[(size_t)p * 65536u + k * 256u + n];
}

// ---- kernel 4: fused per-pair double GEMM + epilogue ---------------------
__global__ __launch_bounds__(256) void qent_pair_kernel(
    const _Float16* __restrict__ Xh,
    const _Float16* __restrict__ WPk,
    const float* __restrict__ b1,
    const float* __restrict__ b2,
    const float* __restrict__ ents,
    const float* __restrict__ SL,
    float* __restrict__ out) {
  __shared__ __align__(16) _Float16 ldsT[64 * XS];       // X tile, then H tile
  __shared__ float rowSq[64];

  const int p    = blockIdx.y;
  const int mb0  = blockIdx.x * 64;                      // batch-row base
  const int li   = cII[p], lj = cJJ[p];
  const int tid  = threadIdx.x;
  const int wave = tid >> 5, lane = tid & 31;

  // load X tile: rows = batch slice, cols 0..127 from layer i, 128..255 from j
  for (int c = tid; c < 2048; c += 256) {
    int row  = c >> 5;
    int colh = (c & 31) << 3;
    int layer = (colh < ND) ? li : lj;
    const v8h* src = (const v8h*)(Xh + ((size_t)(layer*NB + mb0 + row))*ND + (colh & (ND-1)));
    *(v8h*)(&ldsT[row * XS + colh]) = *src;
  }
  if (tid < 64) rowSq[tid] = 0.f;
  __syncthreads();

  const int mb     = wave & 3;                 // wave's 16-row block (0..3)
  const int nb0    = (wave >> 2) << 3;         // wave's 8 N-tiles start (0 or 8)
  const int mrow   = (mb << 4) + (lane & 15);  // A-operand row in tile
  const int koff   = (lane < 16) ? 0 : 8;      // A-fragment K split
  const int ncol   = lane & 15;                // C/D column within 16
  const int mrbase = (lane < 16) ? 0 : 8;      // C/D row base within 16

  const v8f vzero = {0.f,0.f,0.f,0.f,0.f,0.f,0.f,0.f};
  v8f acc[8];
  #pragma unroll
  for (int t = 0; t < 8; ++t) acc[t] = vzero;

  // ================= stage 1: H = tanh(X @ W1 + b1) =================
  {
    const _Float16* wp = WPk + (size_t)(p * 2 + 0) * 65536;
    #pragma unroll
    for (int kb = 0; kb < 8; ++kb) {
      V16u a;
      const _Float16* arow = &ldsT[mrow * XS + (kb << 5) + koff];
      a.h[0] = *(const v8h*)(arow);
      a.h[1] = *(const v8h*)(arow + 16);
      #pragma unroll
      for (int t = 0; t < 8; ++t) {
        int nb = nb0 + t;
        const _Float16* bp = wp + (((kb << 4) + nb) << 9) + (lane << 3);
        V16u bf;
        bf.h[0] = *(const v8h*)(bp);
        bf.h[1] = *(const v8h*)(bp + 256);
        acc[t] = __builtin_amdgcn_wmma_f32_16x16x32_f16(
            false, a.v, false, bf.v, (short)0, acc[t], false, false);
      }
    }
  }
  __syncthreads();  // all waves done reading X tile before we overwrite with H

  #pragma unroll
  for (int t = 0; t < 8; ++t) {
    int nb = nb0 + t;
    float bias = b1[p * TWO_D + (nb << 4) + ncol];
    #pragma unroll
    for (int v = 0; v < 8; ++v) {
      float hv = tanhf(acc[t][v] + bias);
      int m = (mb << 4) + mrbase + v;
      ldsT[m * XS + (nb << 4) + ncol] = (_Float16)hv;
    }
    acc[t] = vzero;
  }
  __syncthreads();

  // ================= stage 2: Bell = H @ W2 + b2 =================
  {
    const _Float16* wp = WPk + (size_t)(p * 2 + 1) * 65536;
    #pragma unroll
    for (int kb = 0; kb < 8; ++kb) {
      V16u a;
      const _Float16* arow = &ldsT[mrow * XS + (kb << 5) + koff];
      a.h[0] = *(const v8h*)(arow);
      a.h[1] = *(const v8h*)(arow + 16);
      #pragma unroll
      for (int t = 0; t < 8; ++t) {
        int nb = nb0 + t;
        const _Float16* bp = wp + (((kb << 4) + nb) << 9) + (lane << 3);
        V16u bf;
        bf.h[0] = *(const v8h*)(bp);
        bf.h[1] = *(const v8h*)(bp + 256);
        acc[t] = __builtin_amdgcn_wmma_f32_16x16x32_f16(
            false, a.v, false, bf.v, (short)0, acc[t], false, false);
      }
    }
  }

  // ================= epilogue: ent, update, mutual info =================
  float strength = 1.f / (1.f + __expf(-ents[li * NL + lj]));
  #pragma unroll
  for (int t = 0; t < 8; ++t) {
    int nb  = nb0 + t;
    int col = (nb << 4) + ncol;
    float bias = b2[p * TWO_D + col];
    int layer  = (col < ND) ? li : lj;
    // closed form of sequential u += alpha*(e - u): coef = 0.1 * 0.9^(7-m)
    float coef = (col < ND) ? 0.1f * cPOW9[7 - lj] : 0.1f * cPOW9[6 - li];
    int dcol   = col & (ND - 1);
    #pragma unroll
    for (int v = 0; v < 8; ++v) {
      int m = (mb << 4) + mrbase + v;
      int b = mb0 + m;
      float bell = acc[t][v] + bias;
      float e = strength * bell
              + 0.01f * gaussn((uint32_t)((uint32_t)(p * NB + b) << 8) | (uint32_t)col);
      atomicAdd(&rowSq[m], e * e);
      atomicAdd(&out[((size_t)layer * NB + b) * ND + dcol], coef * e);
    }
  }
  __syncthreads();

  if (tid < 64) {
    float sq  = rowSq[tid];
    float dn  = fmaxf(sqrtf(sq), 1e-12f);
    float lam = fmaxf(sq / (dn * dn), 1e-12f);
    float S12 = -lam * __logf(lam) + 255.f * 1e-12f * 27.631021115928547f;
    int b = mb0 + tid;
    out[(size_t)UPD_ELEMS + (size_t)p * NB + b] = SL[li * NB + b] + SL[lj * NB + b] - S12;
  }
}

// ---------------------------------------------------------------------------
extern "C" void kernel_launch(void* const* d_in, const int* in_sizes, int n_in,
                              void* d_out, int out_size, void* d_ws, size_t ws_size,
                              hipStream_t stream) {
  (void)in_sizes; (void)n_in; (void)out_size; (void)ws_size;

  const float* states = (const float*)d_in[0];  // [8,8192,128]
  const float* ents   = (const float*)d_in[1];  // [8,8]
  const float* W1     = (const float*)d_in[2];  // [28,256,256]
  const float* b1     = (const float*)d_in[3];  // [28,256]
  const float* W2     = (const float*)d_in[4];  // [28,256,256]
  const float* b2     = (const float*)d_in[5];  // [28,256]
  float* out = (float*)d_out;                   // updated(8.39M) ++ measures(229K)

  char* ws = (char*)d_ws;
  _Float16* Xh  = (_Float16*)ws;                                   // 16 MiB f16 states
  _Float16* WPk = (_Float16*)(ws + (size_t)16777216);              // 7 MiB packed weights
  float*    SL  = (float*)  (ws + (size_t)16777216 + 7340032);     // 256 KiB entropy table

  prep_states <<<8192, 256, 0, stream>>>(states, out, Xh);
  prep_entropy<<< 256, 256, 0, stream>>>(states, SL);
  pack_weights<<<14336, 256, 0, stream>>>(W1, W2, WPk);
  qent_pair_kernel<<<dim3(NB/64, NP), 256, 0, stream>>>(Xh, WPk, b1, b2, ents, SL, out);
}
